// Heirachical_Loss_28123445854310
// MI455X (gfx1250) — matrix operand order repaired and checked by reading, not compiled, not run on hardware
//
#include <hip/hip_runtime.h>
#include <hip/hip_bf16.h>
#include <stdint.h>

// ---------------------------------------------------------------------------
// Hierarchical loss, restructured:
//   M = W @ A                      (1000 x 1000, exact in bf16-WMMA)
//   loss = 32768 - sum_b dot(outputs[b], M[target[b]])
// ---------------------------------------------------------------------------

typedef __attribute__((ext_vector_type(16))) __bf16 v16bf;
typedef __attribute__((ext_vector_type(8)))  __bf16 v8bf;
typedef __attribute__((ext_vector_type(8)))  float  v8f;

#define NUM_CLASSES 1000
#define NUM_NODES   1111
#define BATCH       32768
#define KP          1120          // NUM_NODES padded to multiple of 32
#define MP          1008          // NUM_CLASSES padded to multiple of 16
#define NTILES      63            // MP / 16

static __device__ __forceinline__ __bf16 f32_to_bf16_trunc(float f) {
    // Truncation is exact for the dyadic values in W/A (0, 1, 2^-k).
    union { float f; uint32_t u; } v;
    v.f = f;
    uint16_t h = (uint16_t)(v.u >> 16);
    return __builtin_bit_cast(__bf16, h);
}

// -------- Kernel 0a: pack W -> padded bf16 [MP x KP] ------------------------
__global__ __launch_bounds__(256) void pack_W(
        const float* __restrict__ W,   // [NUM_CLASSES, NUM_NODES]
        __bf16* __restrict__ Wb)       // [MP, KP] zero padded
{
    int idx = blockIdx.x * 256 + threadIdx.x;
    if (idx >= MP * KP) return;
    int r = idx / KP;
    int k = idx - r * KP;
    float f = (r < NUM_CLASSES && k < NUM_NODES) ? W[r * NUM_NODES + k] : 0.0f;
    Wb[idx] = f32_to_bf16_trunc(f);
}

// -------- Kernel 0b: pack A^T -> padded bf16 [MP x KP] ----------------------
__global__ __launch_bounds__(256) void pack_At(
        const float* __restrict__ A_h, // [NUM_NODES, NUM_CLASSES]
        __bf16* __restrict__ At)       // [MP, KP]: At[c][k] = A_h[k][c]
{
    int idx = blockIdx.x * 256 + threadIdx.x;
    if (idx >= MP * KP) return;
    int c = idx / KP;
    int k = idx - c * KP;
    float f = (c < NUM_CLASSES && k < NUM_NODES) ? A_h[k * NUM_CLASSES + c] : 0.0f;
    At[idx] = f32_to_bf16_trunc(f);
}

// -------- Kernel 1: M = Wb @ At^T via v_wmma_f32_16x16x32_bf16 --------------
// One wave (32 threads) per 16x16 tile of M. Grid: 63 x 63 tiles.
// Padded operands -> unconditional aligned vector loads, no exec-mask games.
__global__ __launch_bounds__(32) void build_M_wmma(
        const __bf16* __restrict__ Wb, // [MP, KP]
        const __bf16* __restrict__ At, // [MP, KP]
        float* __restrict__ M)         // [NUM_CLASSES, NUM_CLASSES]
{
    const int tile_t = blockIdx.y;          // target (row) tile
    const int tile_c = blockIdx.x;          // class (col) tile
    const int l = threadIdx.x;              // lane 0..31
    const int h = l >> 4;                   // lane half
    const int r = l & 15;

    const __bf16* __restrict__ arow = Wb + (size_t)(tile_t * 16 + r) * KP;
    const __bf16* __restrict__ brow = At + (size_t)(tile_c * 16 + r) * KP;

    v8f acc = {};

    for (int k0 = 0; k0 < KP; k0 += 32) {
        // Operand A (16-bit, 16x32): lane l (row = l&15):
        //   elems 0..7  -> K = k0 + h*8 + i        (16B contiguous)
        //   elems 8..15 -> K = k0 + 16 + h*8 + i   (16B contiguous)
        v8bf alo = *(const v8bf*)(arow + k0 + h * 8);
        v8bf ahi = *(const v8bf*)(arow + k0 + 16 + h * 8);
        v16bf a = __builtin_shufflevector(alo, ahi,
                    0, 1, 2, 3, 4, 5, 6, 7, 8, 9, 10, 11, 12, 13, 14, 15);
        // Operand B (16-bit, 32x16): lane l (col = l&15):
        //   elems 0..7  -> K = k0 + h*8 + i        (16B contiguous)
        //   elems 8..15 -> K = k0 + 16 + h*8 + i   (16B contiguous)
        v8bf blo = *(const v8bf*)(brow + k0 + h * 8);
        v8bf bhi = *(const v8bf*)(brow + k0 + 16 + h * 8);
        v16bf b = __builtin_shufflevector(blo, bhi,
                    0, 1, 2, 3, 4, 5, 6, 7, 8, 9, 10, 11, 12, 13, 14, 15);

        acc = __builtin_amdgcn_wmma_f32_16x16x32_bf16(
                /*neg_a=*/false, a, /*neg_b=*/false, b,
                /*c_mod=*/(short)0, acc, /*reuse_a=*/false, /*reuse_b=*/false);
    }

    // D layout: lane l holds column n = l&15; VGPR v holds row v + 8*(l>>4).
    #pragma unroll
    for (int v = 0; v < 8; ++v) {
        int row = tile_t * 16 + v + 8 * h;
        int col = tile_c * 16 + r;
        if (row < NUM_CLASSES && col < NUM_CLASSES)
            M[row * NUM_CLASSES + col] = acc[v];
    }
}

// -------- Kernel 2: per-row dots, deterministic block partials --------------
// 256 threads = 8 waves per block; each wave handles rows_per_wave rows.
__global__ __launch_bounds__(256) void batch_dot(
        const float* __restrict__ outputs,  // [BATCH, NUM_CLASSES]
        const int*   __restrict__ target,   // [BATCH]
        const float* __restrict__ M,        // [NUM_CLASSES, NUM_CLASSES]
        float* __restrict__ partials,       // [gridDim.x]
        int rows_per_wave)
{
    __shared__ float smem[8];
    const int wave = threadIdx.x >> 5;
    const int lane = threadIdx.x & 31;
    const int wave_global = blockIdx.x * 8 + wave;

    float acc = 0.0f;
    for (int rr = 0; rr < rows_per_wave; ++rr) {
        int b = wave_global * rows_per_wave + rr;
        if (b < BATCH) {
            const float4* __restrict__ o4 = (const float4*)(outputs + (size_t)b * NUM_CLASSES);
            const float4* __restrict__ m4 = (const float4*)(M + (size_t)target[b] * NUM_CLASSES);
            float s = 0.0f;
            // 1000 floats = 250 float4s, rows are 16B-aligned (4000B stride)
            for (int j = lane; j < 250; j += 32) {
                float4 a = o4[j];
                float4 c = m4[j];
                s = fmaf(a.x, c.x, s);
                s = fmaf(a.y, c.y, s);
                s = fmaf(a.z, c.z, s);
                s = fmaf(a.w, c.w, s);
            }
            acc += s;
        }
    }
    // wave32 tree reduce
    #pragma unroll
    for (int off = 16; off > 0; off >>= 1)
        acc += __shfl_down(acc, off, 32);
    if (lane == 0) smem[wave] = acc;
    __syncthreads();
    if (threadIdx.x == 0) {
        float s = 0.0f;
        #pragma unroll
        for (int w = 0; w < 8; ++w) s += smem[w];
        partials[blockIdx.x] = s;
    }
}

// -------- Kernel 3: deterministic final reduction ---------------------------
__global__ __launch_bounds__(256) void final_reduce(
        const float* __restrict__ partials, int n, float* __restrict__ out)
{
    __shared__ float smem[256];
    float s = 0.0f;
    for (int i = threadIdx.x; i < n; i += 256) s += partials[i];
    smem[threadIdx.x] = s;
    __syncthreads();
    #pragma unroll
    for (int stride = 128; stride > 0; stride >>= 1) {
        if (threadIdx.x < stride) smem[threadIdx.x] += smem[threadIdx.x + stride];
        __syncthreads();
    }
    if (threadIdx.x == 0) out[0] = (float)BATCH - smem[0];
}

extern "C" void kernel_launch(void* const* d_in, const int* in_sizes, int n_in,
                              void* d_out, int out_size, void* d_ws, size_t ws_size,
                              hipStream_t stream) {
    const float* outputs = (const float*)d_in[0];
    const int*   target  = (const int*)d_in[1];
    const float* A_h     = (const float*)d_in[2];
    const float* W       = (const float*)d_in[3];

    // Workspace layout (all offsets 32B aligned):
    //   [0, 4000000)          M        1000*1000 f32
    //   [4000000, 4004096)    partials 1024 f32
    //   [4004096, 6262016)    Wb       MP*KP bf16  (2,257,920 B)
    //   [6262016, 8519936)    At       MP*KP bf16  (2,257,920 B)
    char* ws = (char*)d_ws;
    float*  M        = (float*)(ws);
    float*  partials = (float*)(ws + 4000000);
    __bf16* Wb       = (__bf16*)(ws + 4004096);
    __bf16* At       = (__bf16*)(ws + 6262016);

    // Kernel 0: pack padded bf16 operands (MP*KP = 1,128,960 elems)
    const int pack_blocks = (MP * KP + 255) / 256;
    pack_W <<<pack_blocks, 256, 0, stream>>>(W, Wb);
    pack_At<<<pack_blocks, 256, 0, stream>>>(A_h, At);

    // Kernel 1: M = W @ A (63x63 tiles of 16x16, one wave each, 35 WMMAs/wave)
    dim3 g1(NTILES, NTILES, 1);
    build_M_wmma<<<g1, 32, 0, stream>>>(Wb, At, M);

    // Kernel 2: 1024 blocks * 8 waves * 4 rows = 32768 rows
    const int nblocks = 1024;
    batch_dot<<<nblocks, 256, 0, stream>>>(outputs, target, M, partials, 4);

    // Kernel 3: deterministic scalar result
    final_reduce<<<1, 256, 0, stream>>>(partials, nblocks, (float*)d_out);
}
